// sLSTM_33767032881370
// MI455X (gfx1250) — matrix operand order repaired and checked by reading, not compiled, not run on hardware
//
#include <hip/hip_runtime.h>

// ---------------------------------------------------------------------------
// Types for CDNA5 WMMA (wave32, v_wmma_f32_16x16x32_bf16)
// ---------------------------------------------------------------------------
typedef __attribute__((ext_vector_type(16))) __bf16 v16bf;
typedef __attribute__((ext_vector_type(8)))  __bf16 v8bf;
typedef __attribute__((ext_vector_type(8)))  float  v8f;

#define EPSV 1e-6f

static __device__ __forceinline__ v16bf cat8(v8bf lo, v8bf hi) {
    return __builtin_shufflevector(lo, hi, 0,1,2,3,4,5,6,7,8,9,10,11,12,13,14,15);
}

static __device__ __forceinline__ v8f wmma_bf16(v16bf a, v16bf b, v8f c) {
    // (neg_a, A, neg_b, B, c_mod, C, reuse_a, reuse_b)
    return __builtin_amdgcn_wmma_f32_16x16x32_bf16(false, a, false, b, (short)0, c,
                                                   false, false);
}

// ---------------------------------------------------------------------------
// Convert f32 row-major [rows][cols] -> bf16 transposed [cols][rows]
// ---------------------------------------------------------------------------
__global__ __launch_bounds__(256) void cvt_transpose_kernel(
    const float* __restrict__ src, __bf16* __restrict__ dst, int rows, int cols) {
    __shared__ float tile[32][33];
    const int cb = blockIdx.x * 32;
    const int rb = blockIdx.y * 32;
    const int tx = threadIdx.x;   // 0..31
    const int ty = threadIdx.y;   // 0..7
    #pragma unroll
    for (int i = 0; i < 32; i += 8)
        tile[ty + i][tx] = src[(size_t)(rb + ty + i) * cols + cb + tx];
    __syncthreads();
    #pragma unroll
    for (int i = 0; i < 32; i += 8)
        dst[(size_t)(cb + ty + i) * rows + rb + tx] = (__bf16)tile[tx][ty + i];
}

// ---------------------------------------------------------------------------
// f32 -> bf16 elementwise convert
// ---------------------------------------------------------------------------
__global__ __launch_bounds__(256) void cvt_f32_bf16_kernel(
    const float* __restrict__ s, __bf16* __restrict__ d, size_t n) {
    size_t i = (size_t)blockIdx.x * 256 + threadIdx.x;
    if (i < n) d[i] = (__bf16)s[i];
}

// ---------------------------------------------------------------------------
// Row LayerNorm (no bias) -> bf16.  One block per row, 256 threads.
// ---------------------------------------------------------------------------
__global__ __launch_bounds__(256) void ln_rows_kernel(
    const float* __restrict__ x, const float* __restrict__ scale,
    __bf16* __restrict__ out, int C) {
    __shared__ float sA[256];
    __shared__ float sB[256];
    const int row = blockIdx.x;
    const int tid = threadIdx.x;
    const float* xr = x + (size_t)row * C;
    float s = 0.f, q = 0.f;
    for (int i = tid; i < C; i += 256) { float v = xr[i]; s += v; q += v * v; }
    sA[tid] = s; sB[tid] = q;
    __syncthreads();
    for (int off = 128; off > 0; off >>= 1) {
        if (tid < off) { sA[tid] += sA[tid + off]; sB[tid] += sB[tid + off]; }
        __syncthreads();
    }
    const float mu  = sA[0] / (float)C;
    const float var = sB[0] / (float)C - mu * mu;
    const float inv = rsqrtf(var + EPSV);
    __bf16* orow = out + (size_t)row * C;
    for (int i = tid; i < C; i += 256)
        orow[i] = (__bf16)((xr[i] - mu) * inv * scale[i]);
}

// ---------------------------------------------------------------------------
// WMMA GEMM:  C[M][N] = A_bf16[M][K] * Bt_bf16[N][K]^T (+ bias[N])
//                      (+ A2[M][4*hd] blockdiag R2t[4][hd][hd])  (+ resid[M][N])
// 128x128 block tile, BK=32, 256 threads = 8 waves.
// Wave tile 32x64 (2x4 WMMA tiles): 12 ds_load_b128 per 8 WMMAs, all fragment
// loads hoisted ahead of the WMMA burst.  LDS double-buffered; next K-chunk's
// global loads issue before the compute block to overlap HBM with the XDL pipe.
// ---------------------------------------------------------------------------
__global__ __launch_bounds__(256) void gemm_bt_kernel(
    const __bf16* __restrict__ A,  int lda,
    const __bf16* __restrict__ Bt, int ldb,
    const float*  __restrict__ bias,
    const __bf16* __restrict__ A2, int lda2,
    const __bf16* __restrict__ R2t, int hd,
    const float*  __restrict__ resid,
    float* __restrict__ C, int M, int N, int K) {
    __shared__ __align__(16) __bf16 As[2][128 * 32];
    __shared__ __align__(16) __bf16 Bs[2][128 * 32];

    const int tid     = threadIdx.x;
    const int lane    = tid & 31;
    const int wave    = tid >> 5;           // 0..7
    const int l16     = lane & 15;
    const int halfsel = lane >> 4;          // 0 or 1
    const int wr      = (wave & 3) * 32;    // wave row offset in block tile
    const int wc      = (wave >> 2) * 64;   // wave col offset in block tile
    const int rowBase = blockIdx.y * 128;
    const int colBase = blockIdx.x * 128;

    const v8f acczero = {0.f,0.f,0.f,0.f,0.f,0.f,0.f,0.f};
    v8f acc[8];                              // [i*4+j], i=row tile (2), j=col tile (4)
    #pragma unroll
    for (int t = 0; t < 8; ++t) acc[t] = acczero;

    const int stage_r = tid >> 1;           // 0..127
    const int stage_h = (tid & 1) * 16;     // 0 or 16 (bf16 elements)

    const int nphase = (R2t != nullptr) ? 2 : 1;
    for (int phase = 0; phase < nphase; ++phase) {
        const __bf16* pa;
        const __bf16* pb;
        int plda, pldb, klen;
        if (phase == 0) {
            pa = A  + (size_t)rowBase * lda;  plda = lda;
            pb = Bt + (size_t)colBase * ldb;  pldb = ldb;
            klen = K;
        } else {
            const int g = colBase / hd;       // group index (tile never straddles)
            pa = A2  + (size_t)rowBase * lda2 + (size_t)g * hd;               plda = lda2;
            pb = R2t + (size_t)g * hd * hd + (size_t)(colBase - g * hd) * hd; pldb = hd;
            klen = hd;
        }
        const int nchunk = klen / 32;

        // ---- prologue: stage chunk 0 into buffer 0 ----
        v8bf ra0, ra1, rb0, rb1;
        {
            const __bf16* gA = pa + (size_t)stage_r * plda + stage_h;
            ra0 = ((const v8bf*)gA)[0];  ra1 = ((const v8bf*)gA)[1];
            const __bf16* gB = pb + (size_t)stage_r * pldb + stage_h;
            rb0 = ((const v8bf*)gB)[0];  rb1 = ((const v8bf*)gB)[1];
        }
        __syncthreads();   // phase boundary: previous reads done before overwrite
        *(v8bf*)(&As[0][stage_r * 32 + stage_h    ]) = ra0;
        *(v8bf*)(&As[0][stage_r * 32 + stage_h + 8]) = ra1;
        *(v8bf*)(&Bs[0][stage_r * 32 + stage_h    ]) = rb0;
        *(v8bf*)(&Bs[0][stage_r * 32 + stage_h + 8]) = rb1;

        for (int c = 0; c < nchunk; ++c) {
            const int buf = c & 1;
            const bool hasnext = (c + 1 < nchunk);
            __syncthreads();                  // staged buf visible to all

            if (hasnext) {                    // issue next chunk's global loads now
                const int kk = (c + 1) * 32;
                const __bf16* gA = pa + (size_t)stage_r * plda + kk + stage_h;
                ra0 = ((const v8bf*)gA)[0];  ra1 = ((const v8bf*)gA)[1];
                const __bf16* gB = pb + (size_t)stage_r * pldb + kk + stage_h;
                rb0 = ((const v8bf*)gB)[0];  rb1 = ((const v8bf*)gB)[1];
                if (c + 2 < nchunk) {         // global_prefetch_b8 two chunks ahead
                    __builtin_prefetch(pa + (size_t)stage_r * plda + kk + 32, 0, 1);
                    __builtin_prefetch(pb + (size_t)stage_r * pldb + kk + 32, 0, 1);
                }
            }

            // ---- load ALL fragments first, then a back-to-back WMMA burst ----
            const __bf16* Abuf = As[buf];
            const __bf16* Bbuf = Bs[buf];
            v16bf afrag[2];
            #pragma unroll
            for (int i = 0; i < 2; ++i) {
                // A 16x32 bf16 ISA layout: lanes<16 K={0..7,16..23}, lanes>=16 +8
                const int arow = wr + i * 16 + l16;
                v8bf a0 = *(const v8bf*)(&Abuf[arow * 32 + halfsel * 8]);
                v8bf a1 = *(const v8bf*)(&Abuf[arow * 32 + 16 + halfsel * 8]);
                afrag[i] = cat8(a0, a1);
            }
            v16bf bfrag[4];
            #pragma unroll
            for (int j = 0; j < 4; ++j) {
                // B 32x16 bf16 layout: lanes<16 K={0..7,16..23}, lanes>=16 +8
                const int bcol = wc + j * 16 + l16;
                v8bf b0 = *(const v8bf*)(&Bbuf[bcol * 32 + halfsel * 8]);
                v8bf b1 = *(const v8bf*)(&Bbuf[bcol * 32 + 16 + halfsel * 8]);
                bfrag[j] = cat8(b0, b1);
            }
            #pragma unroll
            for (int i = 0; i < 2; ++i)
                #pragma unroll
                for (int j = 0; j < 4; ++j)
                    acc[i * 4 + j] = wmma_bf16(afrag[i], bfrag[j], acc[i * 4 + j]);

            if (hasnext) {                    // write next chunk into other buffer
                *(v8bf*)(&As[buf ^ 1][stage_r * 32 + stage_h    ]) = ra0;
                *(v8bf*)(&As[buf ^ 1][stage_r * 32 + stage_h + 8]) = ra1;
                *(v8bf*)(&Bs[buf ^ 1][stage_r * 32 + stage_h    ]) = rb0;
                *(v8bf*)(&Bs[buf ^ 1][stage_r * 32 + stage_h + 8]) = rb1;
            }
        }
    }

    // Epilogue: C/D layout — VGPR r, lanes0-15: (M=r, N=lane); lanes16-31: (M=8+r)
    #pragma unroll
    for (int j = 0; j < 4; ++j) {
        const int coln = colBase + wc + j * 16 + l16;
        const float badd = bias ? bias[coln] : 0.f;
        #pragma unroll
        for (int i = 0; i < 2; ++i) {
            #pragma unroll
            for (int r = 0; r < 8; ++r) {
                const int rown = rowBase + wr + i * 16 + r + halfsel * 8;
                float v = acc[i * 4 + j][r] + badd;
                if (resid) v += resid[(size_t)rown * N + coln];
                C[(size_t)rown * N + coln] = v;
            }
        }
    }
}

// ---------------------------------------------------------------------------
// sLSTM gates + GroupNorm(4 groups) + residual + LN2 -> y_bf16.
// One block per row (C=2048), 256 threads, 8 elements/thread (group-aligned).
// ---------------------------------------------------------------------------
__global__ __launch_bounds__(256) void slstm_elem_kernel(
    const float* __restrict__ zp, const float* __restrict__ ip,
    const float* __restrict__ fp, const float* __restrict__ op,
    const float* __restrict__ seq,
    const float* __restrict__ c_old, const float* __restrict__ n_old,
    const float* __restrict__ m_old,
    const float* __restrict__ gn_s, const float* __restrict__ ln2_s,
    float* __restrict__ out_c, float* __restrict__ out_n,
    float* __restrict__ out_h, float* __restrict__ out_m,
    __bf16* __restrict__ y_bf, int C) {
    __shared__ float sA[256];
    __shared__ float sB[256];
    const int row = blockIdx.x;
    const int tid = threadIdx.x;
    const size_t base = (size_t)row * C + tid * 8;

    float hv[8];
    float gs = 0.f, gq = 0.f;
    #pragma unroll
    for (int e = 0; e < 8; ++e) {
        const size_t i = base + e;
        const float z  = tanhf(zp[i]);
        const float iv = ip[i];
        const float fv = fp[i];
        const float mo = m_old[i];
        const float mn = fmaxf(fv + mo, iv);
        const float ig = expf(iv - mn);
        const float fg = expf(fv - mn + mo);
        const float og = 1.f / (1.f + expf(-op[i]));
        const float ct = fg * c_old[i] + ig * z;
        const float nt = fg * n_old[i] + ig;
        const float ht = og * (ct / nt);
        out_c[i] = ct; out_n[i] = nt; out_h[i] = ht; out_m[i] = mn;
        hv[e] = ht; gs += ht; gq += ht * ht;
    }
    // group reduce over 64-thread segments (512 channels / group)
    sA[tid] = gs; sB[tid] = gq;
    __syncthreads();
    for (int off = 32; off > 0; off >>= 1) {
        if ((tid & 63) < off) { sA[tid] += sA[tid + off]; sB[tid] += sB[tid + off]; }
        __syncthreads();
    }
    const int segbase = tid & ~63;
    const float gmu  = sA[segbase] / 512.f;
    const float gvar = sB[segbase] / 512.f - gmu * gmu;
    const float ginv = rsqrtf(gvar + EPSV);
    __syncthreads();

    float ov[8];
    float rs = 0.f, rq = 0.f;
    #pragma unroll
    for (int e = 0; e < 8; ++e) {
        const size_t i = base + e;
        const int ch = tid * 8 + e;
        const float v = (hv[e] - gmu) * ginv * gn_s[ch] + seq[i];
        ov[e] = v; rs += v; rq += v * v;
    }
    sA[tid] = rs; sB[tid] = rq;
    __syncthreads();
    for (int off = 128; off > 0; off >>= 1) {
        if (tid < off) { sA[tid] += sA[tid + off]; sB[tid] += sB[tid + off]; }
        __syncthreads();
    }
    const float mu2  = sA[0] / (float)C;
    const float var2 = sB[0] / (float)C - mu2 * mu2;
    const float inv2 = rsqrtf(var2 + EPSV);
    #pragma unroll
    for (int e = 0; e < 8; ++e) {
        const int ch = tid * 8 + e;
        y_bf[base + e] = (__bf16)((ov[e] - mu2) * inv2 * ln2_s[ch]);
    }
}

// ---------------------------------------------------------------------------
// silu(gate) * up, then LN over F with mlp_ln_s -> bf16.
// One block per row (F=8192), 256 threads, 32 elements/thread.
// ---------------------------------------------------------------------------
__global__ __launch_bounds__(256) void swiglu_ln_kernel(
    const float* __restrict__ gate, const float* __restrict__ up,
    const float* __restrict__ mlp_ln_s, __bf16* __restrict__ mid, int F) {
    __shared__ float sA[256];
    __shared__ float sB[256];
    const int row = blockIdx.x;
    const int tid = threadIdx.x;
    const size_t base = (size_t)row * F + tid * 32;

    float v[32];
    float s = 0.f, q = 0.f;
    #pragma unroll
    for (int e = 0; e < 32; ++e) {
        const size_t i = base + e;
        const float g = gate[i];
        const float u = up[i];
        const float sg = g / (1.f + expf(-g));   // silu
        const float val = sg * u;
        v[e] = val; s += val; q += val * val;
    }
    sA[tid] = s; sB[tid] = q;
    __syncthreads();
    for (int off = 128; off > 0; off >>= 1) {
        if (tid < off) { sA[tid] += sA[tid + off]; sB[tid] += sB[tid + off]; }
        __syncthreads();
    }
    const float mu  = sA[0] / (float)F;
    const float var = sB[0] / (float)F - mu * mu;
    const float inv = rsqrtf(var + EPSV);
    #pragma unroll
    for (int e = 0; e < 32; ++e)
        mid[base + e] = (__bf16)((v[e] - mu) * inv * mlp_ln_s[tid * 32 + e]);
}

// ---------------------------------------------------------------------------
// Host orchestration
// ---------------------------------------------------------------------------
extern "C" void kernel_launch(void* const* d_in, const int* in_sizes, int n_in,
                              void* d_out, int out_size, void* d_ws, size_t ws_size,
                              hipStream_t stream) {
    (void)in_sizes; (void)n_in; (void)out_size; (void)ws_size;
    const int B = 4096, C = 2048, F = 8192, HD = 512;

    const float* seq    = (const float*)d_in[0];
    const float* c_tm1  = (const float*)d_in[1];
    const float* n_tm1  = (const float*)d_in[2];
    const float* h_tm1  = (const float*)d_in[3];
    const float* m_tm1  = (const float*)d_in[4];
    const float* ln1_s  = (const float*)d_in[5];
    const float* Ws[4]  = {(const float*)d_in[6], (const float*)d_in[8],
                           (const float*)d_in[10], (const float*)d_in[12]};
    const float* bs[4]  = {(const float*)d_in[7], (const float*)d_in[9],
                           (const float*)d_in[11], (const float*)d_in[13]};
    const float* Rs[4]  = {(const float*)d_in[14], (const float*)d_in[15],
                           (const float*)d_in[16], (const float*)d_in[17]};
    const float* gn_s   = (const float*)d_in[18];
    const float* ln2_s  = (const float*)d_in[19];
    const float* gate_k = (const float*)d_in[20];
    const float* up_k   = (const float*)d_in[21];
    const float* mlp_ln = (const float*)d_in[22];
    const float* down_k = (const float*)d_in[23];

    // workspace layout
    char* ws = (char*)d_ws;
    size_t off = 0;
    auto wsalloc = [&](size_t bytes) -> char* {
        char* p = ws + off;
        off += (bytes + 255) & ~(size_t)255;
        return p;
    };
    __bf16* WT[4];
    for (int g = 0; g < 4; ++g) WT[g] = (__bf16*)wsalloc((size_t)C * C * 2);
    __bf16* RT     = (__bf16*)wsalloc((size_t)16 * HD * HD * 2);  // [gate][blk][n][k]
    __bf16* gateT  = (__bf16*)wsalloc((size_t)F * C * 2);
    __bf16* upT    = (__bf16*)wsalloc((size_t)F * C * 2);
    __bf16* downT  = (__bf16*)wsalloc((size_t)C * F * 2);
    __bf16* x_bf   = (__bf16*)wsalloc((size_t)B * C * 2);
    __bf16* h_bf   = (__bf16*)wsalloc((size_t)B * C * 2);
    __bf16* y_bf   = (__bf16*)wsalloc((size_t)B * C * 2);
    __bf16* mid_bf = (__bf16*)wsalloc((size_t)B * F * 2);
    float* pre[4];
    for (int g = 0; g < 4; ++g) pre[g] = (float*)wsalloc((size_t)B * C * 4);
    float* gate_f = (float*)wsalloc((size_t)B * F * 4);
    float* up_f   = (float*)wsalloc((size_t)B * F * 4);

    float* out0  = (float*)d_out;
    float* out_c = out0 + (size_t)B * C;
    float* out_n = out0 + (size_t)2 * B * C;
    float* out_h = out0 + (size_t)3 * B * C;
    float* out_m = out0 + (size_t)4 * B * C;

    const dim3 tb(32, 8);
    // weight convert + transpose (f32 -> bf16, [K][N] -> [N][K])
    for (int g = 0; g < 4; ++g)
        cvt_transpose_kernel<<<dim3(C / 32, C / 32), tb, 0, stream>>>(Ws[g], WT[g], C, C);
    for (int g = 0; g < 4; ++g)
        for (int h = 0; h < 4; ++h)
            cvt_transpose_kernel<<<dim3(HD / 32, HD / 32), tb, 0, stream>>>(
                Rs[g] + (size_t)h * HD * HD, RT + (size_t)(g * 4 + h) * HD * HD, HD, HD);
    cvt_transpose_kernel<<<dim3(F / 32, C / 32), tb, 0, stream>>>(gate_k, gateT, C, F);
    cvt_transpose_kernel<<<dim3(F / 32, C / 32), tb, 0, stream>>>(up_k, upT, C, F);
    cvt_transpose_kernel<<<dim3(C / 32, F / 32), tb, 0, stream>>>(down_k, downT, F, C);

    // x_t = LN(seq)*ln1_s -> bf16 ;  h_tm1 -> bf16
    ln_rows_kernel<<<B, 256, 0, stream>>>(seq, ln1_s, x_bf, C);
    cvt_f32_bf16_kernel<<<(B * C + 255) / 256, 256, 0, stream>>>(
        h_tm1, h_bf, (size_t)B * C);

    // z/i/f/o preacts: x@W + bias + h@blockdiag(R)
    for (int g = 0; g < 4; ++g)
        gemm_bt_kernel<<<dim3(C / 128, B / 128), 256, 0, stream>>>(
            x_bf, C, WT[g], C, bs[g],
            h_bf, C, RT + (size_t)g * 4 * HD * HD, HD,
            nullptr, pre[g], B, C, C);

    // sLSTM cell + GroupNorm + residual + LN2 -> y_bf16 (and c/n/h/m outputs)
    slstm_elem_kernel<<<B, 256, 0, stream>>>(
        pre[0], pre[1], pre[2], pre[3], seq, c_tm1, n_tm1, m_tm1,
        gn_s, ln2_s, out_c, out_n, out_h, out_m, y_bf, C);

    // MLP gate / up projections
    gemm_bt_kernel<<<dim3(F / 128, B / 128), 256, 0, stream>>>(
        y_bf, C, gateT, C, nullptr, nullptr, 0, nullptr, 0, nullptr,
        gate_f, B, F, C);
    gemm_bt_kernel<<<dim3(F / 128, B / 128), 256, 0, stream>>>(
        y_bf, C, upT, C, nullptr, nullptr, 0, nullptr, 0, nullptr,
        up_f, B, F, C);

    // silu(gate)*up + LN -> mid_bf16
    swiglu_ln_kernel<<<B, 256, 0, stream>>>(gate_f, up_f, mlp_ln, mid_bf, F);

    // down projection + seq residual -> d_out[0 : B*C]
    gemm_bt_kernel<<<dim3(C / 128, B / 128), 256, 0, stream>>>(
        mid_bf, F, downT, F, nullptr, nullptr, 0, nullptr, 0, seq,
        out0, B, C, F);
}
